// PINNModel_44281112822143
// MI455X (gfx1250) — matrix operand order repaired and checked by reading, not compiled, not run on hardware
//
#include <hip/hip_runtime.h>
#include <cstddef>

// ---------------------------------------------------------------------------
// PINN residual R = phi_tt - alpha^2 (phi_xx + phi_zz)
// phi diag-Hessian via forward 2nd-order jets, batched 16 points/wave onto
// v_wmma_f32_16x16x32_f16 (f16 mul, f32 acc). Compute-bound (~29 GFLOP vs 4MB).
// Activations use branchless hw-transcendental tanh (v_exp_f32 + v_rcp_f32)
// so the VALU side co-executes with the XDL WMMA pipe instead of dominating.
// ---------------------------------------------------------------------------

typedef _Float16 v8h  __attribute__((ext_vector_type(8)));
typedef _Float16 v16h __attribute__((ext_vector_type(16)));
typedef float    v8f  __attribute__((ext_vector_type(8)));

#define WAVES_PER_BLOCK 2
#define BLOCK_THREADS   64
// per-wave LDS: 2 x 7 x (16x64) f16 jet buffers + 2 x (16x32) f16 alpha state
#define WAVE_LDS_BYTES  (2 * 7 * 2048 + 2 * 1024)   // 30720

// workspace layout (bytes)
#define PHI_WT_OFF 0u          // 5 layers x 64x64 f16 (n-major, k contiguous)
#define PHI_B_OFF  40960u      // 5 x 64 f32
#define ALP_WT_OFF 42240u      // 6 layers x 32x32 f16 (n-major)
#define ALP_B_OFF  54528u      // 6 x 32 f32
#define WS_BYTES   55296u

struct PrepArgs {
    const float* phiW[5];
    const float* phiBp[5];
    const float* alpW[6];
    const float* alpBp[6];
};

// -------------------------- weight prep kernel ------------------------------
__global__ void pinn_prep_kernel(PrepArgs pa, char* __restrict__ ws) {
    _Float16* phiWt = (_Float16*)(ws + PHI_WT_OFF);
    float*    phiB  = (float*)   (ws + PHI_B_OFF);
    _Float16* alpWt = (_Float16*)(ws + ALP_WT_OFF);
    float*    alpB  = (float*)   (ws + ALP_B_OFF);

    const int phi_in[5]  = {3, 50, 50, 50, 50};
    const int phi_out[5] = {50, 50, 50, 50, 1};
    const int alp_in[6]  = {2, 20, 20, 20, 20, 20};
    const int alp_out[6] = {20, 20, 20, 20, 20, 1};

    const int tid    = threadIdx.x + blockIdx.x * blockDim.x;
    const int stride = blockDim.x * gridDim.x;

    // phi weights transposed+padded: phiWt[l][n][k]
    for (int idx = tid; idx < 5 * 64 * 64; idx += stride) {
        int l = idx >> 12, r = idx & 4095, n = r >> 6, k = r & 63;
        float v = (n < phi_out[l] && k < phi_in[l])
                    ? pa.phiW[l][k * phi_out[l] + n] : 0.0f;
        phiWt[idx] = (_Float16)v;
    }
    for (int idx = tid; idx < 5 * 64; idx += stride) {
        int l = idx >> 6, n = idx & 63;
        phiB[idx] = (n < phi_out[l]) ? pa.phiBp[l][n] : 0.0f;
    }
    // alpha weights transposed+padded: alpWt[l][n][k], 32x32
    for (int idx = tid; idx < 6 * 32 * 32; idx += stride) {
        int l = idx >> 10, r = idx & 1023, n = r >> 5, k = r & 31;
        float v = (n < alp_out[l] && k < alp_in[l])
                    ? pa.alpW[l][k * alp_out[l] + n] : 0.0f;
        alpWt[idx] = (_Float16)v;
    }
    for (int idx = tid; idx < 6 * 32; idx += stride) {
        int l = idx >> 5, n = idx & 31;
        alpB[idx] = (n < alp_out[l]) ? pa.alpBp[l][n] : 0.0f;
    }
}

// --------------------------- fragment helpers -------------------------------
// A operand (16x32 f16, ISA 7.12.2): lane = row M (mod 16);
// lanes 0-15 hold K 0..7 & 16..23, lanes 16-31 hold K 8..15 & 24..31.
__device__ __forceinline__ v16h load_a_frag(const _Float16* buf, int ld,
                                            int kt, int lane) {
    const int M  = lane & 15;
    const int kb = (lane < 16) ? 0 : 8;
    const _Float16* p = buf + M * ld + (kt << 5) + kb;
    v8h lo = *(const v8h*)(p);
    v8h hi = *(const v8h*)(p + 16);
    return __builtin_shufflevector(lo, hi, 0,1,2,3,4,5,6,7,
                                           8,9,10,11,12,13,14,15);
}

// B operand (32x16 f16): lane = column N (mod 16);
// lanes 0-15 hold K 0..7 & 16..23, lanes 16-31 hold K 8..15 & 24..31.
// Weights stored n-major (k contiguous) so this is two 16B loads.
__device__ __forceinline__ v16h load_b_frag(const _Float16* wt, int ld,
                                            int nt, int kt, int lane) {
    const int N  = (nt << 4) + (lane & 15);
    const int kb = (lane < 16) ? 0 : 8;
    const _Float16* p = wt + N * ld + (kt << 5) + kb;
    v8h lo = *(const v8h*)(p);
    v8h hi = *(const v8h*)(p + 16);
    return __builtin_shufflevector(lo, hi, 0,1,2,3,4,5,6,7,
                                           8,9,10,11,12,13,14,15);
}

__device__ __forceinline__ v8f wmma_f16(v16h a, v16h b, v8f c) {
    return __builtin_amdgcn_wmma_f32_16x16x32_f16(
        false, a, false, b, (short)0, c, false, false);
}

// Branchless tanh: 1 - 2/(exp(2u)+1).  v_exp_f32 + v_rcp_f32, correct +-1
// saturation for |u| -> inf, error far below the f16 storage quantization.
__device__ __forceinline__ float fast_tanh(float u) {
    float e = __expf(2.0f * u);
    return 1.0f - 2.0f * __builtin_amdgcn_rcpf(e + 1.0f);
}

__device__ __forceinline__ float sigf(float t) {
    return __builtin_amdgcn_rcpf(1.0f + __expf(-t));
}

// ------------------------------ main kernel ---------------------------------
__global__ __launch_bounds__(BLOCK_THREADS)
void pinn_residual_kernel(const float* __restrict__ X,
                          const char* __restrict__ ws,
                          float* __restrict__ out, int npts) {
    extern __shared__ char smem[];

    const int lane = threadIdx.x & 31;
    const int wave = threadIdx.x >> 5;
    const int tile = blockIdx.x * WAVES_PER_BLOCK + wave;
    const int base = tile << 4;
    if (base >= npts) return;                 // wave-uniform

    const _Float16* phiWt = (const _Float16*)(ws + PHI_WT_OFF);
    const float*    phiB  = (const float*)   (ws + PHI_B_OFF);
    const _Float16* alpWt = (const _Float16*)(ws + ALP_WT_OFF);
    const float*    alpB  = (const float*)   (ws + ALP_B_OFF);

    // keep small weight set hot in cache (global_prefetch_b8)
    for (unsigned o = (unsigned)threadIdx.x * 128u; o < WS_BYTES;
         o += (unsigned)BLOCK_THREADS * 128u)
        __builtin_prefetch(ws + o, 0, 3);

    // wave-private LDS: ping-pong jet buffers (no barriers needed, same-wave
    // DS ops are in order)
    char* wb = smem + wave * WAVE_LDS_BYTES;
    _Float16* Q[2][7];
    #pragma unroll
    for (int b = 0; b < 2; ++b)
        #pragma unroll
        for (int q = 0; q < 7; ++q)
            Q[b][q] = (_Float16*)(wb + (b * 7 + q) * 2048);
    _Float16* A[2] = { (_Float16*)(wb + 14 * 2048),
                       (_Float16*)(wb + 14 * 2048 + 1024) };

    // input scaling h0 = 2x/UB - 1, UB = {2,2,1} -> seed = {1,1,2}
    const float sc[3] = {1.0f, 1.0f, 2.0f};

    // ---- initialize jets (q0 = value, q1..3 = d/dx_i, q4..6 = d2/dx_i^2) ----
    for (int e = lane; e < 16 * 64; e += 32) {
        int m = e >> 6, k = e & 63;
        float v = 0.0f;
        if (k < 3 && (base + m) < npts)
            v = sc[k] * X[(size_t)(base + m) * 3 + k] - 1.0f;
        Q[0][0][e] = (_Float16)v;
        #pragma unroll
        for (int i = 0; i < 3; ++i) {
            Q[0][1 + i][e] = (_Float16)((k == i) ? sc[i] : 0.0f);
            Q[0][4 + i][e] = (_Float16)0.0f;
        }
    }
    // alpha input: 2*xz/UB_ALP - 1 = xz - 1
    for (int e = lane; e < 16 * 32; e += 32) {
        int m = e >> 5, k = e & 31;
        float v = 0.0f;
        if (k < 2 && (base + m) < npts)
            v = X[(size_t)(base + m) * 3 + k] - 1.0f;
        A[0][e] = (_Float16)v;
    }

    const int col    = lane & 15;
    const int rowoff = (lane < 16) ? 0 : 8;

    // --------------------- phi: 4 tanh layers of jets -----------------------
    #pragma unroll
    for (int l = 0; l < 4; ++l) {
        const _Float16* Wt = phiWt + l * 4096;
        const float*    bl = phiB + l * 64;
        _Float16* const* in  = Q[l & 1];
        _Float16* const* qo  = Q[(l + 1) & 1];

        #pragma unroll
        for (int nt = 0; nt < 4; ++nt) {
            v16h b0 = load_b_frag(Wt, 64, nt, 0, lane);
            v16h b1 = load_b_frag(Wt, 64, nt, 1, lane);
            v8f acc[7];
            #pragma unroll
            for (int q = 0; q < 7; ++q) {
                v16h a0 = load_a_frag(in[q], 64, 0, lane);
                v16h a1 = load_a_frag(in[q], 64, 1, lane);
                v8f c = {};
                c = wmma_f16(a0, b0, c);
                c = wmma_f16(a1, b1, c);
                acc[q] = c;
            }
            const float bias = bl[(nt << 4) + col];
            #pragma unroll
            for (int g = 0; g < 8; ++g) {
                float u   = acc[0][g] + bias;
                float t   = fast_tanh(u);
                float tp  = 1.0f - t * t;
                float tpp = -2.0f * t * tp;
                int idx = (rowoff + g) * 64 + (nt << 4) + col;
                qo[0][idx] = (_Float16)t;
                #pragma unroll
                for (int i = 0; i < 3; ++i) {
                    float du  = acc[1 + i][g];
                    float d2u = acc[4 + i][g];
                    qo[1 + i][idx] = (_Float16)(tp * du);
                    qo[4 + i][idx] = (_Float16)(tpp * du * du + tp * d2u);
                }
            }
        }
    }
    // final jet state lives in Q[0]

    // ------------------------ alpha: 5 tanh layers --------------------------
    #pragma unroll
    for (int l = 0; l < 5; ++l) {
        const _Float16* Wt = alpWt + l * 1024;
        const float*    bl = alpB + l * 32;
        const _Float16* in = A[l & 1];
        _Float16*       ao = A[(l + 1) & 1];
        #pragma unroll
        for (int nt = 0; nt < 2; ++nt) {
            v16h b = load_b_frag(Wt, 32, nt, 0, lane);
            v16h a = load_a_frag(in, 32, 0, lane);
            v8f c = {};
            c = wmma_f16(a, b, c);
            const float bias = bl[(nt << 4) + col];
            #pragma unroll
            for (int g = 0; g < 8; ++g) {
                float t = fast_tanh(c[g] + bias);
                ao[(rowoff + g) * 32 + (nt << 4) + col] = (_Float16)t;
            }
        }
    }
    // final alpha hidden state lives in A[1]

    // ------- final linear layers + mask + residual (lanes 0..15 = points) ---
    if (lane < 16) {
        const int m = lane;
        if (base + m < npts) {
            const _Float16* w4 = phiWt + 4 * 4096;   // row n=0, k contiguous
            float sxx = 0.0f, szz = 0.0f, stt = 0.0f;
            #pragma unroll
            for (int k = 0; k < 64; ++k) {
                float w = (float)w4[k];
                sxx += (float)Q[0][4][m * 64 + k] * w;
                szz += (float)Q[0][5][m * 64 + k] * w;
                stt += (float)Q[0][6][m * 64 + k] * w;
            }
            const _Float16* w5 = alpWt + 5 * 1024;
            float sa = 0.0f;
            #pragma unroll
            for (int k = 0; k < 32; ++k)
                sa += (float)A[1][m * 32 + k] * (float)w5[k];
            sa += alpB[5 * 32 + 0];
            float astar = fast_tanh(sa);

            float xx = X[(size_t)(base + m) * 3 + 0];
            float zz = X[(size_t)(base + m) * 3 + 1];
            float mask = sigf(1000.0f * (zz - 0.5f)) *
                         sigf(1000.0f * (1.5f - zz)) *
                         sigf(1000.0f * (xx - 0.5f)) *
                         sigf(1000.0f * (1.5f - xx));
            float alpha = 3.0f + 2.0f * astar * mask;
            out[base + m] = stt - alpha * alpha * (sxx + szz);
        }
    }
}

// ------------------------------- launcher -----------------------------------
extern "C" void kernel_launch(void* const* d_in, const int* in_sizes, int n_in,
                              void* d_out, int out_size, void* d_ws,
                              size_t ws_size, hipStream_t stream) {
    (void)n_in; (void)out_size; (void)ws_size;
    const float* X = (const float*)d_in[0];
    PrepArgs pa;
    for (int l = 0; l < 5; ++l) {
        pa.phiW[l]  = (const float*)d_in[1 + 2 * l];
        pa.phiBp[l] = (const float*)d_in[2 + 2 * l];
    }
    for (int l = 0; l < 6; ++l) {
        pa.alpW[l]  = (const float*)d_in[11 + 2 * l];
        pa.alpBp[l] = (const float*)d_in[12 + 2 * l];
    }
    char* ws = (char*)d_ws;
    pinn_prep_kernel<<<32, 256, 0, stream>>>(pa, ws);

    const int B      = in_sizes[0] / 3;
    const int tiles  = (B + 15) / 16;
    const int blocks = (tiles + WAVES_PER_BLOCK - 1) / WAVES_PER_BLOCK;
    pinn_residual_kernel<<<blocks, BLOCK_THREADS,
                           WAVES_PER_BLOCK * WAVE_LDS_BYTES, stream>>>(
        X, ws, (float*)d_out, B);
}
